// TauGridNetwork_57458072486273
// MI455X (gfx1250) — matrix operand order repaired and checked by reading, not compiled, not run on hardware
//
#include <hip/hip_runtime.h>

#define T_STEPS 100
#define B_SZ    256
#define NP_SZ   512
#define NG_SZ   4096
#define SKP     40   // padded LDS row stride (32 data bf16 + 8 pad) -> 80B, conflict-free

typedef __bf16 bf16_t;
typedef __attribute__((ext_vector_type(16))) __bf16 v16bf;
typedef __attribute__((ext_vector_type(8)))  float  v8f;
typedef int v4i_vs __attribute__((vector_size(16)));   // matches builtin param type

union FragU { v16bf v; uint4 u[2]; unsigned w[8]; };

#define AS1 __attribute__((address_space(1)))
#define AS3 __attribute__((address_space(3)))

// CDNA5 async global->LDS copy path (ASYNCcnt), with manual fallback.
#if defined(__has_builtin)
#if __has_builtin(__builtin_amdgcn_global_load_async_to_lds_b128) && \
    __has_builtin(__builtin_amdgcn_s_wait_asynccnt)
#define USE_ASYNC_LDS 1
#endif
#endif

// ---------------- fp32 -> bf16 helpers -------------------------------------
__device__ __forceinline__ unsigned short f2bf(float f) {
    unsigned u = __float_as_uint(f);
    unsigned r = (u + 0x7FFFu + ((u >> 16) & 1u)) >> 16;
    return (unsigned short)r;
}
__device__ __forceinline__ unsigned pk2(float lo, float hi) {
#if __has_builtin(__builtin_amdgcn_cvt_pk_bf16_f32)
    auto r = __builtin_amdgcn_cvt_pk_bf16_f32(lo, hi);
    union { decltype(r) v; unsigned u; } cv;
    cv.v = r;
    return cv.u;
#else
    return (unsigned)f2bf(lo) | ((unsigned)f2bf(hi) << 16);
#endif
}

// ---------------- WMMA fragment loaders (ISA 7.12.2, wave32) ---------------
__device__ __forceinline__ v16bf frag_a_g_bf16(const bf16_t* A, int lda,
                                               int row, int k0, int lane) {
    int kb = ((lane >> 4) & 1) * 8;
    const bf16_t* p = A + (size_t)row * lda + k0 + kb;
    FragU fu;
    fu.u[0] = *(const uint4*)(p);
    fu.u[1] = *(const uint4*)(p + 16);
    return fu.v;
}
__device__ __forceinline__ v16bf frag_a_g_f32(const float* A, int lda,
                                              int row, int k0, int lane) {
    int kb = ((lane >> 4) & 1) * 8;
    const float* p = A + (size_t)row * lda + k0 + kb;
    float4 f0 = *(const float4*)(p + 0);
    float4 f1 = *(const float4*)(p + 4);
    float4 f2 = *(const float4*)(p + 16);
    float4 f3 = *(const float4*)(p + 20);
    FragU fu;
    fu.w[0] = pk2(f0.x, f0.y); fu.w[1] = pk2(f0.z, f0.w);
    fu.w[2] = pk2(f1.x, f1.y); fu.w[3] = pk2(f1.z, f1.w);
    fu.w[4] = pk2(f2.x, f2.y); fu.w[5] = pk2(f2.z, f2.w);
    fu.w[6] = pk2(f3.x, f3.y); fu.w[7] = pk2(f3.z, f3.w);
    return fu.v;
}
__device__ __forceinline__ v16bf frag_a_lds(const bf16_t* la, int moff, int lane) {
    int m  = (lane & 15) + moff;
    int kb = ((lane >> 4) & 1) * 8;
    const bf16_t* p = la + m * SKP + kb;
    FragU fu;
    fu.u[0] = *(const uint4*)(p);
    fu.u[1] = *(const uint4*)(p + 16);
    return fu.v;
}
__device__ __forceinline__ v16bf frag_b_g_bf16(const bf16_t* Bm, int ldb,
                                               int col0, int k0, int lane) {
    int n  = lane & 15;
    int kb = ((lane >> 4) & 1) * 8;
    const bf16_t* p = Bm + (size_t)(col0 + n) * ldb + k0 + kb;
    FragU fu;
    fu.u[0] = *(const uint4*)(p);
    fu.u[1] = *(const uint4*)(p + 16);
    return fu.v;
}
__device__ __forceinline__ v16bf frag_b_lds(const bf16_t* lb, int tile, int lane) {
    int n  = (tile * 16) + (lane & 15);
    int kb = ((lane >> 4) & 1) * 8;
    const bf16_t* p = lb + n * SKP + kb;
    FragU fu;
    fu.u[0] = *(const uint4*)(p);
    fu.u[1] = *(const uint4*)(p + 16);
    return fu.v;
}

// Stage one 16B segment of the B strip into LDS (async if available).
__device__ __forceinline__ void stage_b128(const bf16_t* gsrc, bf16_t* ldst) {
#ifdef USE_ASYNC_LDS
    __builtin_amdgcn_global_load_async_to_lds_b128(
        (AS1 v4i_vs*)gsrc, (AS3 v4i_vs*)ldst, 0, 0);
#else
    *(uint4*)ldst = *(const uint4*)gsrc;
#endif
}
__device__ __forceinline__ void stage_drain() {
#ifdef USE_ASYNC_LDS
    __builtin_amdgcn_s_wait_asynccnt(0);
#endif
}

// ---------------- prep kernels ---------------------------------------------
__global__ void cvt_bf16_kernel(const float* __restrict__ src,
                                unsigned short* __restrict__ dst, int n) {
    int i = blockIdx.x * blockDim.x + threadIdx.x;
    int stride = gridDim.x * blockDim.x;
    for (; i < n; i += stride) dst[i] = f2bf(src[i]);
}

__global__ void alpha_kernel(const float* __restrict__ tau,
                             float* __restrict__ alpha, int n) {
    int i = blockIdx.x * blockDim.x + threadIdx.x;
    if (i < n) {
        float t = fminf(fmaxf(tau[i], 1.0f), 50.0f);
        alpha[i] = 1.0f / t;
    }
}

// ---------------- h0 = init_pc @ enc_W^T  (256 x 4096, K=512) --------------
__global__ void __launch_bounds__(256)
h0_kernel(const float* __restrict__ init_pc,   // (B, NP)
          const bf16_t* __restrict__ encb,     // (NG, NP)
          float* __restrict__ h_out,           // (B, NG)
          bf16_t* __restrict__ h_out_b) {
    __shared__ bf16_t lb[2][64 * SKP];
    const int tid  = threadIdx.x;
    const int lane = tid & 31;
    const int wave = tid >> 5;
    const int nsb  = NG_SZ / 64;
    const int mblk = blockIdx.x / nsb;
    const int ns   = blockIdx.x % nsb;
    const int col0 = ns * 64;
    const int row0 = (mblk * 8 + wave) * 16;
    const int arow = row0 + (lane & 15);

    const int scol = tid >> 2;
    const int sseg = (tid & 3) * 8;
    const int soff = scol * SKP + sseg;
    const bf16_t* gsrc = encb + (size_t)(col0 + scol) * NP_SZ + sseg;

    stage_b128(gsrc, &lb[0][soff]);
    stage_drain();
    __syncthreads();

    v8f acc[4] = {};
    for (int k0 = 0; k0 < NP_SZ; k0 += 32) {
        const int cur = (k0 >> 5) & 1;
        const bool more = (k0 + 32 < NP_SZ);
#ifdef USE_ASYNC_LDS
        if (more) stage_b128(gsrc + k0 + 32, &lb[cur ^ 1][soff]);
#else
        uint4 stg = {};
        if (more) stg = *(const uint4*)(gsrc + k0 + 32);
#endif
        v16bf a  = frag_a_g_f32(init_pc, NP_SZ, arow, k0, lane);
        v16bf b0 = frag_b_lds(lb[cur], 0, lane);
        v16bf b1 = frag_b_lds(lb[cur], 1, lane);
        v16bf b2 = frag_b_lds(lb[cur], 2, lane);
        v16bf b3 = frag_b_lds(lb[cur], 3, lane);
        acc[0] = __builtin_amdgcn_wmma_f32_16x16x32_bf16(false, a, false, b0, (short)0, acc[0], false, false);
        acc[1] = __builtin_amdgcn_wmma_f32_16x16x32_bf16(false, a, false, b1, (short)0, acc[1], false, false);
        acc[2] = __builtin_amdgcn_wmma_f32_16x16x32_bf16(false, a, false, b2, (short)0, acc[2], false, false);
        acc[3] = __builtin_amdgcn_wmma_f32_16x16x32_bf16(false, a, false, b3, (short)0, acc[3], false, false);
#ifdef USE_ASYNC_LDS
        if (more) stage_drain();
#else
        if (more) *(uint4*)&lb[cur ^ 1][soff] = stg;
#endif
        __syncthreads();
    }
    const int m_off = ((lane >> 4) & 1) * 8;
    unsigned short* hob = (unsigned short*)h_out_b;
#pragma unroll
    for (int j = 0; j < 4; ++j) {
        int g = col0 + 16 * j + (lane & 15);
#pragma unroll
        for (int r = 0; r < 8; ++r) {
            int bi = row0 + m_off + r;
            float hv = acc[j][r];
            size_t idx = (size_t)bi * NG_SZ + g;
            h_out[idx] = hv;
            hob[idx]   = f2bf(hv);
        }
    }
}

// ---------------- one recurrent step ---------------------------------------
// pre = v . W_ih + h @ W_hh^T ; h' = h + alpha*(relu(pre) - h)
__global__ void __launch_bounds__(256)
step_kernel(const float*  __restrict__ h_in,    // (B, NG) fp32
            const bf16_t* __restrict__ h_in_b,  // (B, NG) bf16
            const bf16_t* __restrict__ whhb,    // (NG, NG) bf16, L2-resident
            const float*  __restrict__ vel_t,   // (B, 2)
            const float*  __restrict__ W_ih,    // (NG, 2)
            const float*  __restrict__ alpha,   // (NG)
            float*  __restrict__ h_out,
            bf16_t* __restrict__ h_out_b,
            float*  __restrict__ g_out) {       // (B, NG)
    __shared__ bf16_t lb[2][64 * SKP];
    const int tid  = threadIdx.x;
    const int lane = tid & 31;
    const int wave = tid >> 5;
    const int nsb  = NG_SZ / 64;
    const int mblk = blockIdx.x / nsb;
    const int ns   = blockIdx.x % nsb;
    const int col0 = ns * 64;
    const int row0 = (mblk * 8 + wave) * 16;
    const int arow = row0 + (lane & 15);

    const int scol = tid >> 2;
    const int sseg = (tid & 3) * 8;
    const int soff = scol * SKP + sseg;
    const bf16_t* gsrc = whhb + (size_t)(col0 + scol) * NG_SZ + sseg;

    stage_b128(gsrc, &lb[0][soff]);
    stage_drain();
    __syncthreads();

    v8f acc[4] = {};
    v16bf a_cur = frag_a_g_bf16(h_in_b, NG_SZ, arow, 0, lane);
    for (int k0 = 0; k0 < NG_SZ; k0 += 32) {
        const int cur = (k0 >> 5) & 1;
        const bool more = (k0 + 32 < NG_SZ);
#ifdef USE_ASYNC_LDS
        if (more) stage_b128(gsrc + k0 + 32, &lb[cur ^ 1][soff]);
#else
        uint4 stg = {};
        if (more) stg = *(const uint4*)(gsrc + k0 + 32);
#endif
        v16bf a_nxt = a_cur;
        if (more) a_nxt = frag_a_g_bf16(h_in_b, NG_SZ, arow, k0 + 32, lane);
        v16bf b0 = frag_b_lds(lb[cur], 0, lane);
        v16bf b1 = frag_b_lds(lb[cur], 1, lane);
        v16bf b2 = frag_b_lds(lb[cur], 2, lane);
        v16bf b3 = frag_b_lds(lb[cur], 3, lane);
        acc[0] = __builtin_amdgcn_wmma_f32_16x16x32_bf16(false, a_cur, false, b0, (short)0, acc[0], false, false);
        acc[1] = __builtin_amdgcn_wmma_f32_16x16x32_bf16(false, a_cur, false, b1, (short)0, acc[1], false, false);
        acc[2] = __builtin_amdgcn_wmma_f32_16x16x32_bf16(false, a_cur, false, b2, (short)0, acc[2], false, false);
        acc[3] = __builtin_amdgcn_wmma_f32_16x16x32_bf16(false, a_cur, false, b3, (short)0, acc[3], false, false);
        a_cur = a_nxt;
#ifdef USE_ASYNC_LDS
        if (more) stage_drain();
#else
        if (more) *(uint4*)&lb[cur ^ 1][soff] = stg;
#endif
        __syncthreads();
    }
    const int m_off = ((lane >> 4) & 1) * 8;
    unsigned short* hob = (unsigned short*)h_out_b;
#pragma unroll
    for (int j = 0; j < 4; ++j) {
        int g = col0 + 16 * j + (lane & 15);
        float w0 = W_ih[g * 2 + 0];
        float w1 = W_ih[g * 2 + 1];
        float al = alpha[g];
#pragma unroll
        for (int r = 0; r < 8; ++r) {
            int bi = row0 + m_off + r;
            float vx = vel_t[bi * 2 + 0];
            float vy = vel_t[bi * 2 + 1];
            float pre = acc[j][r] + vx * w0 + vy * w1;
            float rl  = pre > 0.0f ? pre : 0.0f;
            size_t idx = (size_t)bi * NG_SZ + g;
            float hv = h_in[idx];
            float hn = hv + al * (rl - hv);
            h_out[idx] = hn;
            hob[idx]   = f2bf(hn);
            g_out[idx] = hn;
        }
    }
}

// ---------------- place_logits = g @ dec_W^T  (25600 x 512, K=4096) --------
__global__ void __launch_bounds__(256)
decode_kernel(const float*  __restrict__ g,     // (T*B, NG) fp32
              const bf16_t* __restrict__ decb,  // (NP, NG) bf16, L2-resident
              float* __restrict__ out) {        // (T*B, NP)
    __shared__ bf16_t la[2][32 * SKP];
    const int tid  = threadIdx.x;
    const int lane = tid & 31;
    const int wave = tid >> 5;
    const int row0 = blockIdx.x * 32;
    const int col0 = wave * 64;

    const int srow = tid >> 3;                   // 0..31
    const int sseg = (tid & 7) * 4;              // 0,4,..,28
    const float* gsrc = g + (size_t)(row0 + srow) * NG_SZ + sseg;

    {
        float4 d = *(const float4*)(gsrc);
        unsigned* dst = (unsigned*)&la[0][srow * SKP + sseg];
        dst[0] = pk2(d.x, d.y);
        dst[1] = pk2(d.z, d.w);
    }
    __syncthreads();

    v8f acc[2][4] = {};
    for (int k0 = 0; k0 < NG_SZ; k0 += 32) {
        const int cur = (k0 >> 5) & 1;
        const bool more = (k0 + 32 < NG_SZ);
        float4 stg = {};
        if (more) stg = *(const float4*)(gsrc + k0 + 32);   // load early ...
        v16bf a0 = frag_a_lds(la[cur], 0, lane);
        v16bf a1 = frag_a_lds(la[cur], 16, lane);
        v16bf b0 = frag_b_g_bf16(decb, NG_SZ, col0 + 0,  k0, lane);
        v16bf b1 = frag_b_g_bf16(decb, NG_SZ, col0 + 16, k0, lane);
        v16bf b2 = frag_b_g_bf16(decb, NG_SZ, col0 + 32, k0, lane);
        v16bf b3 = frag_b_g_bf16(decb, NG_SZ, col0 + 48, k0, lane);
        acc[0][0] = __builtin_amdgcn_wmma_f32_16x16x32_bf16(false, a0, false, b0, (short)0, acc[0][0], false, false);
        acc[1][0] = __builtin_amdgcn_wmma_f32_16x16x32_bf16(false, a1, false, b0, (short)0, acc[1][0], false, false);
        acc[0][1] = __builtin_amdgcn_wmma_f32_16x16x32_bf16(false, a0, false, b1, (short)0, acc[0][1], false, false);
        acc[1][1] = __builtin_amdgcn_wmma_f32_16x16x32_bf16(false, a1, false, b1, (short)0, acc[1][1], false, false);
        acc[0][2] = __builtin_amdgcn_wmma_f32_16x16x32_bf16(false, a0, false, b2, (short)0, acc[0][2], false, false);
        acc[1][2] = __builtin_amdgcn_wmma_f32_16x16x32_bf16(false, a1, false, b2, (short)0, acc[1][2], false, false);
        acc[0][3] = __builtin_amdgcn_wmma_f32_16x16x32_bf16(false, a0, false, b3, (short)0, acc[0][3], false, false);
        acc[1][3] = __builtin_amdgcn_wmma_f32_16x16x32_bf16(false, a1, false, b3, (short)0, acc[1][3], false, false);
        if (more) {                                          // ... commit late
            unsigned* dst = (unsigned*)&la[cur ^ 1][srow * SKP + sseg];
            dst[0] = pk2(stg.x, stg.y);
            dst[1] = pk2(stg.z, stg.w);
        }
        __syncthreads();
    }
    const int m_off = ((lane >> 4) & 1) * 8;
#pragma unroll
    for (int mi = 0; mi < 2; ++mi) {
#pragma unroll
        for (int j = 0; j < 4; ++j) {
            int p = col0 + 16 * j + (lane & 15);
#pragma unroll
            for (int r = 0; r < 8; ++r) {
                int row = row0 + mi * 16 + m_off + r;
                out[(size_t)row * NP_SZ + p] = acc[mi][j][r];
            }
        }
    }
}

// ---------------- host side ------------------------------------------------
extern "C" void kernel_launch(void* const* d_in, const int* in_sizes, int n_in,
                              void* d_out, int out_size, void* d_ws, size_t ws_size,
                              hipStream_t stream) {
    (void)in_sizes; (void)n_in; (void)out_size; (void)ws_size;
    const float* velocity = (const float*)d_in[0];   // (T, B, 2)
    const float* init_pc  = (const float*)d_in[1];   // (B, NP)
    const float* enc_W    = (const float*)d_in[2];   // (NG, NP)
    const float* dec_W    = (const float*)d_in[3];   // (NP, NG)
    const float* W_ih     = (const float*)d_in[4];   // (NG, 2)
    const float* W_hh     = (const float*)d_in[5];   // (NG, NG)
    const float* tau      = (const float*)d_in[6];   // (NG)

    char* w = (char*)d_ws;
    size_t off = 0;
    auto alloc = [&](size_t bytes) -> void* {
        void* p = w + off;
        off = (off + bytes + 255) & ~(size_t)255;
        return p;
    };
    bf16_t* whh_b = (bf16_t*)alloc((size_t)NG_SZ * NG_SZ * 2);   // 32 MB
    bf16_t* enc_b = (bf16_t*)alloc((size_t)NG_SZ * NP_SZ * 2);   //  4 MB
    bf16_t* dec_b = (bf16_t*)alloc((size_t)NP_SZ * NG_SZ * 2);   //  4 MB
    float*  alpha = (float*) alloc((size_t)NG_SZ * 4);
    float*  h_f[2];
    bf16_t* h_b[2];
    h_f[0] = (float*) alloc((size_t)B_SZ * NG_SZ * 4);
    h_f[1] = (float*) alloc((size_t)B_SZ * NG_SZ * 4);
    h_b[0] = (bf16_t*)alloc((size_t)B_SZ * NG_SZ * 2);
    h_b[1] = (bf16_t*)alloc((size_t)B_SZ * NG_SZ * 2);

    cvt_bf16_kernel<<<2048, 256, 0, stream>>>(W_hh, (unsigned short*)whh_b, NG_SZ * NG_SZ);
    cvt_bf16_kernel<<<512, 256, 0, stream>>>(enc_W, (unsigned short*)enc_b, NG_SZ * NP_SZ);
    cvt_bf16_kernel<<<512, 256, 0, stream>>>(dec_W, (unsigned short*)dec_b, NP_SZ * NG_SZ);
    alpha_kernel<<<(NG_SZ + 255) / 256, 256, 0, stream>>>(tau, alpha, NG_SZ);

    const int gemm_blocks = 2 * (NG_SZ / 64);   // 128
    h0_kernel<<<gemm_blocks, 256, 0, stream>>>(init_pc, enc_b, h_f[0], h_b[0]);

    float* logits = (float*)d_out;                           // (T, B, NP)
    float* gbase  = logits + (size_t)T_STEPS * B_SZ * NP_SZ; // (T, B, NG)

    for (int t = 0; t < T_STEPS; ++t) {
        int rd = t & 1, wr = rd ^ 1;
        step_kernel<<<gemm_blocks, 256, 0, stream>>>(
            h_f[rd], h_b[rd], whh_b,
            velocity + (size_t)t * B_SZ * 2, W_ih, alpha,
            h_f[wr], h_b[wr],
            gbase + (size_t)t * B_SZ * NG_SZ);
    }

    const int dec_blocks = (T_STEPS * B_SZ) / 32;            // 800
    decode_kernel<<<dec_blocks, 256, 0, stream>>>(gbase, dec_b, logits);
}
